// TokenMergingLayer_73529840107771
// MI455X (gfx1250) — matrix operand order repaired and checked by reading, not compiled, not run on hardware
//
#include <hip/hip_runtime.h>
#include <hip/hip_bf16.h>

#define DM    1024
#define NROWS 32768
#define NRED  16384
#define NSAVE 16384
#define MT    128
#define NT    128

typedef __attribute__((ext_vector_type(16))) __bf16 v16bf;
typedef __attribute__((ext_vector_type(8)))  float  v8f;

union FragU { uint4 u[2]; v16bf v; };

// Split two f32 into packed bf16 hi (truncated) and packed bf16 lo.
// hiPk = {bf16(f1), bf16(f0)}, loPk = {bf16(f1-h1), bf16(f0-h0)}
// f - hi is exact (hi is a mantissa prefix of f), so hi+lo carries ~16
// mantissa bits -> bf16x3 WMMA accumulation is ~fp32 quality.
__device__ __forceinline__ void split_pair(float f0, float f1,
                                           unsigned& hiPk, unsigned& loPk) {
    unsigned u0 = __builtin_bit_cast(unsigned, f0);
    unsigned u1 = __builtin_bit_cast(unsigned, f1);
    float h0 = __builtin_bit_cast(float, u0 & 0xFFFF0000u);
    float h1 = __builtin_bit_cast(float, u1 & 0xFFFF0000u);
    hiPk = __builtin_amdgcn_perm(u1, u0, 0x07060302u);   // v_perm_b32
    float l0 = f0 - h0;
    float l1 = f1 - h1;
    loPk = __builtin_amdgcn_perm(__builtin_bit_cast(unsigned, l1),
                                 __builtin_bit_cast(unsigned, l0),
                                 0x07060302u);
}

// ---------------- CSR build kernels ----------------

__global__ void tm_zero_kernel(int* __restrict__ p, int n) {
    int i = blockIdx.x * blockDim.x + threadIdx.x;
    if (i < n) p[i] = 0;
}

__global__ void tm_count_kernel(const int* __restrict__ ids_save,
                                int* __restrict__ counts, int n) {
    int i = blockIdx.x * blockDim.x + threadIdx.x;
    if (i < n) atomicAdd(&counts[ids_save[i]], 1);
}

__global__ __launch_bounds__(1024)
void tm_scan_kernel(const int* __restrict__ counts,
                    int* __restrict__ row_start,
                    int* __restrict__ cursor) {
    __shared__ int sh[1024];
    const int t = threadIdx.x;
    int local[32];
    const int base = t * 32;
    int sum = 0;
#pragma unroll
    for (int k = 0; k < 32; ++k) { local[k] = counts[base + k]; sum += local[k]; }
    sh[t] = sum;
    __syncthreads();
    for (int off = 1; off < 1024; off <<= 1) {
        int v = (t >= off) ? sh[t - off] : 0;
        __syncthreads();
        sh[t] += v;
        __syncthreads();
    }
    int run = sh[t] - sum;   // exclusive prefix
#pragma unroll
    for (int k = 0; k < 32; ++k) {
        row_start[base + k] = run;
        cursor[base + k]    = run;
        run += local[k];
    }
    if (t == 1023) row_start[NROWS] = run;
}

__global__ void tm_fill_kernel(const int* __restrict__ ids_save,
                               int* __restrict__ cursor,
                               int* __restrict__ entries, int n) {
    int i = blockIdx.x * blockDim.x + threadIdx.x;
    if (i < n) {
        int p = atomicAdd(&cursor[ids_save[i]], 1);
        entries[p] = i;
    }
}

// ---------------- out = gather(x, ids_to_save) ----------------

__global__ __launch_bounds__(256)
void tm_gather_init_kernel(const float* __restrict__ x,
                           const int* __restrict__ ids_save,
                           float* __restrict__ out) {
    const size_t idx = (size_t)blockIdx.x * blockDim.x + threadIdx.x;
    const int row = (int)(idx >> 8);          // DM/4 = 256 float4 per row
    const int c   = (int)(idx & 255);
    const float4* src =
        (const float4*)(x + (size_t)ids_save[row] * DM) + c;
    ((float4*)out)[idx] = *src;
}

// ---------------- GEMM (bf16x3 WMMA) + CSR scatter-add ----------------

__global__ __launch_bounds__(256)
void tm_gemm_scatter_kernel(const float* __restrict__ x,
                            const float* __restrict__ W,
                            const int* __restrict__ ids_red,
                            const int* __restrict__ row_start,
                            const int* __restrict__ entries,
                            float* __restrict__ out) {
    // double-buffered bf16 hi/lo tiles, packed 2 bf16 per dword
    __shared__ __attribute__((aligned(16))) unsigned Ahi[2][MT * 16];
    __shared__ __attribute__((aligned(16))) unsigned Alo[2][MT * 16];
    __shared__ __attribute__((aligned(16))) unsigned Bhi[2][NT * 16];
    __shared__ __attribute__((aligned(16))) unsigned Blo[2][NT * 16];
    __shared__ int sh_rid[MT];
    __shared__ int sh_s[MT];
    __shared__ int sh_e[MT];

    const int t     = threadIdx.x;
    const int mBase = blockIdx.x * MT;       // reduce-row block
    const int n0    = blockIdx.y * NT;       // output-column block

    if (t < MT) {
        int rid = ids_red[mBase + t];
        int tgt = rid + 1;
        sh_rid[t] = rid;
        sh_s[t]   = row_start[tgt];
        sh_e[t]   = row_start[tgt + 1];
    }
    __syncthreads();

    const int wave   = t >> 5;
    const int lane   = t & 31;
    const int half   = lane >> 4;            // 0 or 1
    const int l16    = lane & 15;
    const int wave_r = wave >> 1;            // 0..3 -> 32-row slab
    const int wave_c = wave & 1;             // 0..1 -> 64-col slab

    const v8f vzero = {0.f, 0.f, 0.f, 0.f, 0.f, 0.f, 0.f, 0.f};
    v8f acc[2][4];
#pragma unroll
    for (int i = 0; i < 2; ++i)
#pragma unroll
        for (int j = 0; j < 4; ++j) acc[i][j] = vzero;

    // staging assignment: 2 threads per row/col, 16 floats (8 dwords) each
    const int sr  = t >> 1;
    const int sk  = (t & 1) * 16;            // float offset within 32-chunk
    const int sk2 = (t & 1) * 8;             // dword offset
    const float* aSrcBase = x + (size_t)sh_rid[sr] * DM + sk;
    const float* bSrcBase = W + (size_t)(n0 + sr) * DM + sk;

    auto stage = [&](int k0, int buf) {
        const float4* aS = (const float4*)(aSrcBase + k0);
        const float4* bS = (const float4*)(bSrcBase + k0);
        float4 av[4], bv[4];
#pragma unroll
        for (int q = 0; q < 4; ++q) { av[q] = aS[q]; bv[q] = bS[q]; }
        if (k0 + 32 < DM) {
            __builtin_prefetch(aSrcBase + k0 + 32, 0, 0);   // global_prefetch_b8
            __builtin_prefetch(bSrcBase + k0 + 32, 0, 0);
        }
        uint4 ah[2], al[2], bh[2], bl[2];
        unsigned* pah = (unsigned*)ah;
        unsigned* pal = (unsigned*)al;
        unsigned* pbh = (unsigned*)bh;
        unsigned* pbl = (unsigned*)bl;
#pragma unroll
        for (int q = 0; q < 4; ++q) {
            split_pair(av[q].x, av[q].y, pah[2 * q],     pal[2 * q]);
            split_pair(av[q].z, av[q].w, pah[2 * q + 1], pal[2 * q + 1]);
            split_pair(bv[q].x, bv[q].y, pbh[2 * q],     pbl[2 * q]);
            split_pair(bv[q].z, bv[q].w, pbh[2 * q + 1], pbl[2 * q + 1]);
        }
        uint4* dAh = (uint4*)&Ahi[buf][sr * 16 + sk2];
        uint4* dAl = (uint4*)&Alo[buf][sr * 16 + sk2];
        uint4* dBh = (uint4*)&Bhi[buf][sr * 16 + sk2];
        uint4* dBl = (uint4*)&Blo[buf][sr * 16 + sk2];
        dAh[0] = ah[0]; dAh[1] = ah[1];          // ds_store_b128 x8
        dAl[0] = al[0]; dAl[1] = al[1];
        dBh[0] = bh[0]; dBh[1] = bh[1];
        dBl[0] = bl[0]; dBl[1] = bl[1];
    };

    stage(0, 0);
    __syncthreads();

    for (int kk = 0; kk < DM / 32; ++kk) {
        const int buf = kk & 1;
        if (kk + 1 < DM / 32) stage((kk + 1) * 32, buf ^ 1);

        const uint4* pAhi = (const uint4*)Ahi[buf];
        const uint4* pAlo = (const uint4*)Alo[buf];
        const uint4* pBhi = (const uint4*)Bhi[buf];
        const uint4* pBlo = (const uint4*)Blo[buf];

        // fragments per ISA 16-bit A(16x32) / B(32x16) layouts -> ds_load_b128
        FragU aHi[2], aLo[2], bHiF[4], bLoF[4];
#pragma unroll
        for (int rt = 0; rt < 2; ++rt) {
            const int r  = wave_r * 32 + rt * 16 + l16;
            const int c0 = r * 4 + half;          // uint4 index, chunk 0
            aHi[rt].u[0] = pAhi[c0];
            aHi[rt].u[1] = pAhi[c0 + 2];
            aLo[rt].u[0] = pAlo[c0];
            aLo[rt].u[1] = pAlo[c0 + 2];
        }
#pragma unroll
        for (int ct = 0; ct < 4; ++ct) {
            const int n  = wave_c * 64 + ct * 16 + l16;
            const int c0 = n * 4 + half;
            bHiF[ct].u[0] = pBhi[c0];
            bHiF[ct].u[1] = pBhi[c0 + 2];
            bLoF[ct].u[0] = pBlo[c0];
            bLoF[ct].u[1] = pBlo[c0 + 2];
        }

        // bf16x3: hi*hi + hi*lo + lo*hi, f32 accumulate
#pragma unroll
        for (int rt = 0; rt < 2; ++rt)
#pragma unroll
            for (int ct = 0; ct < 4; ++ct) {
                v8f a0 = acc[rt][ct];
                a0 = __builtin_amdgcn_wmma_f32_16x16x32_bf16(
                        false, aHi[rt].v, false, bHiF[ct].v, (short)0, a0, false, false);
                a0 = __builtin_amdgcn_wmma_f32_16x16x32_bf16(
                        false, aHi[rt].v, false, bLoF[ct].v, (short)0, a0, false, false);
                a0 = __builtin_amdgcn_wmma_f32_16x16x32_bf16(
                        false, aLo[rt].v, false, bHiF[ct].v, (short)0, a0, false, false);
                acc[rt][ct] = a0;
            }
        __syncthreads();
    }

    // ---- epilogue: scatter-add into matching output rows via CSR ----
#pragma unroll
    for (int rt = 0; rt < 2; ++rt) {
        const int rowb = wave_r * 32 + rt * 16;
#pragma unroll
        for (int ct = 0; ct < 4; ++ct) {
            const int n = n0 + wave_c * 64 + ct * 16 + l16;
#pragma unroll
            for (int v = 0; v < 8; ++v) {
                const int mloc = rowb + v + 8 * half;
                const float val = acc[rt][ct][v];
                const int s = sh_s[mloc];
                const int e = sh_e[mloc];
                for (int p = s; p < e; ++p) {
                    const int i = entries[p];
                    atomicAdd(out + (size_t)i * DM + n, val);
                }
            }
        }
    }
}

// ---------------- launcher ----------------

extern "C" void kernel_launch(void* const* d_in, const int* in_sizes, int n_in,
                              void* d_out, int out_size, void* d_ws, size_t ws_size,
                              hipStream_t stream) {
    const float* x        = (const float*)d_in[0];
    const float* W        = (const float*)d_in[1];
    const int*   ids_save = (const int*)d_in[2];
    const int*   ids_red  = (const int*)d_in[3];
    float*       out      = (float*)d_out;

    char* ws = (char*)d_ws;
    // ws layout (~459 KB total)
    int* row_start = (int*)(ws + 0);         // NROWS+1 ints
    int* counts    = (int*)(ws + 131328);    // NROWS ints
    int* cursor    = (int*)(ws + 262400);    // NROWS ints
    int* entries   = (int*)(ws + 393472);    // NSAVE ints

    tm_zero_kernel<<<(NROWS + 255) / 256, 256, 0, stream>>>(counts, NROWS);
    tm_count_kernel<<<(NSAVE + 255) / 256, 256, 0, stream>>>(ids_save, counts, NSAVE);
    tm_scan_kernel<<<1, 1024, 0, stream>>>(counts, row_start, cursor);
    tm_fill_kernel<<<(NSAVE + 255) / 256, 256, 0, stream>>>(ids_save, cursor, entries, NSAVE);

    // out = x[ids_to_save]  (one float4 per thread, exact grid)
    tm_gather_init_kernel<<<(NSAVE * (DM / 4)) / 256, 256, 0, stream>>>(x, ids_save, out);

    // GEMM + scatter-add
    dim3 grid(NRED / MT, DM / NT);   // (128, 8)
    tm_gemm_scatter_kernel<<<grid, 256, 0, stream>>>(x, W, ids_red, row_start, entries, out);
}
